// GroupedQueryAttention_26895085208212
// MI455X (gfx1250) — compile-verified, matrix operand order checked by reading, not run on hardware
//
#include <hip/hip_runtime.h>

typedef _Float16 v16h __attribute__((ext_vector_type(16)));
typedef _Float16 v8h  __attribute__((ext_vector_type(8)));
typedef float    v8f  __attribute__((ext_vector_type(8)));

constexpr int BB  = 4;      // batch
constexpr int SEQ = 2048;   // sequence length
constexpr int NH  = 16;     // query heads
constexpr int NKV = 4;      // kv heads
constexpr int HD  = 64;     // head dim
constexpr int HID = 1024;   // hidden

union FragA { v16h v; v8h h[2]; };

static __device__ __forceinline__ v8f wmma_f16(const FragA& a, const FragA& b, v8f c) {
  // D = A(16x32 f16) x B(32x16 f16) + C(16x16 f32)
  return __builtin_amdgcn_wmma_f32_16x16x32_f16(false, a.v, false, b.v, (short)0, c,
                                                false, false);
}

// ---------------------------------------------------------------- converters
__global__ void cvt_f32_f16(const float* __restrict__ in, _Float16* __restrict__ out, int n) {
  int i = blockIdx.x * blockDim.x + threadIdx.x;
  if (i < n) out[i] = (_Float16)in[i];
}

// W [K][N] f32  ->  Wt [N][K] f16
__global__ void transpose_cvt(const float* __restrict__ W, _Float16* __restrict__ Wt,
                              int K, int N) {
  int i = blockIdx.x * blockDim.x + threadIdx.x;
  if (i < K * N) {
    int k = i / N, n = i % N;
    Wt[(size_t)n * K + k] = (_Float16)W[i];
  }
}

// ------------------------------------------------------------------- GEMM
// C[M,N] = A[M,K] (f16, row major) x Bt[N,K]^T (f16, N-major).
// One wave computes a 16(M) x 64(N) strip: 4 accumulators, A frag reused 4x.
// TRANS: store f16 transposed as out[(b*N+n)*Scol + s]  (for V^T)
// F32OUT: store f32 row-major (final projection)
template<bool TRANS, bool F32OUT>
__global__ __launch_bounds__(256) void gemm_wmma(
    const _Float16* __restrict__ A, const _Float16* __restrict__ Bt,
    void* __restrict__ outp, int M, int N, int K, int Scol) {
  const int lane = threadIdx.x & 31;
  // Scalarize the per-wave tile index: proves uniformity -> SALU control flow,
  // SGPR-resident tile bases, EXEC untouched around WMMA.
  const int wave = __builtin_amdgcn_readfirstlane(
      (blockIdx.x * blockDim.x + threadIdx.x) >> 5);
  const int nT64 = N >> 6;
  const int mt   = wave / nT64;
  const int nt64 = wave % nT64;
  const int hi = lane >> 4, lo = lane & 15;

  const v8f vzero = {0.f,0.f,0.f,0.f,0.f,0.f,0.f,0.f};
  v8f acc[4];
#pragma unroll
  for (int t = 0; t < 4; ++t) acc[t] = vzero;

  const _Float16* arow = A + (size_t)(mt * 16 + lo) * K + hi * 8;
  for (int k0 = 0; k0 < K; k0 += 32) {
    FragA a;
    a.h[0] = *(const v8h*)(arow + k0);
    a.h[1] = *(const v8h*)(arow + k0 + 16);
#pragma unroll
    for (int t = 0; t < 4; ++t) {
      const _Float16* brow = Bt + (size_t)(nt64 * 64 + t * 16 + lo) * K + hi * 8;
      FragA b;
      b.h[0] = *(const v8h*)(brow + k0);
      b.h[1] = *(const v8h*)(brow + k0 + 16);
      acc[t] = wmma_f16(a, b, acc[t]);
    }
  }

#pragma unroll
  for (int t = 0; t < 4; ++t) {
    const int n = nt64 * 64 + t * 16 + lo;
    if constexpr (TRANS) {
      const int m0 = mt * 16 + hi * 8;        // 8 consecutive rows -> contiguous s
      const int bb = m0 / Scol;
      const int s0 = m0 % Scol;
      v8h pk;
#pragma unroll
      for (int i = 0; i < 8; ++i) pk[i] = (_Float16)acc[t][i];
      *(v8h*)((_Float16*)outp + ((size_t)(bb * N + n)) * Scol + s0) = pk;
    } else if constexpr (F32OUT) {
      float* o = (float*)outp;
#pragma unroll
      for (int i = 0; i < 8; ++i)
        o[(size_t)(mt * 16 + hi * 8 + i) * N + n] = acc[t][i];
    } else {
      _Float16* o = (_Float16*)outp;
#pragma unroll
      for (int i = 0; i < 8; ++i)
        o[(size_t)(mt * 16 + hi * 8 + i) * N + n] = (_Float16)acc[t][i];
    }
  }
}

// -------------------------------------------------------------------- RoPE
// In-place on Qh [B][S][NH][64] and Kh [B][S][NKV][64]; thread j handles pair (j, j+32).
// Q is additionally pre-scaled by 1/sqrt(HD) so the attention loop needs no scaling.
__global__ void rope_kernel(_Float16* __restrict__ Qh, _Float16* __restrict__ Kh, int total) {
  int tid = blockIdx.x * blockDim.x + threadIdx.x;
  if (tid >= total) return;
  const int j    = tid & 31;
  const int rest = tid >> 5;
  const int hh   = rest % (NH + NKV);
  const int s    = (rest / (NH + NKV)) % SEQ;
  const int b    = rest / ((NH + NKV) * SEQ);
  const bool isQ = (hh < NH);
  _Float16* ptr = isQ
      ? (Qh + (((size_t)(b * SEQ + s)) * NH  + hh)        * HD)
      : (Kh + (((size_t)(b * SEQ + s)) * NKV + (hh - NH)) * HD);
  // inv_freq[j] = 10000^(-2j/64) = exp(-j * ln(10000)/32)
  const float inv = __expf(-(float)j * (9.210340371976184f / 32.0f));
  const float ang = (float)s * inv;
  float sn, cs;
  __sincosf(ang, &sn, &cs);
  const float scl = isQ ? 0.125f : 1.0f;     // fold 1/sqrt(64) into Q
  const float x0 = (float)ptr[j], x1 = (float)ptr[j + 32];
  ptr[j]      = (_Float16)((x0 * cs - x1 * sn) * scl);
  ptr[j + 32] = (_Float16)((x1 * cs + x0 * sn) * scl);
}

// ---------------------------------------------------------- flash attention
// One wave per 16-row Q tile of one (b,h). 32 keys/iter, online softmax.
// Blocks strictly below the diagonal skip masking entirely (scalar branch).
__global__ __launch_bounds__(256) void flash_attn(
    const _Float16* __restrict__ Qh, const _Float16* __restrict__ Kh,
    const _Float16* __restrict__ Vt, _Float16* __restrict__ Oh) {
  __shared__ __align__(16) _Float16 smem[8 * 512];   // 16x32 f16 P buffer per wave
  const int lane  = threadIdx.x & 31;
  const int wslot = threadIdx.x >> 5;
  // Scalarize wave id: q0/nkb/nFull/h/b become SGPRs, loop guards become
  // s_cbranch (EXEC stays all-ones for every WMMA as the ISA requires).
  const int wave = __builtin_amdgcn_readfirstlane(
      (blockIdx.x * blockDim.x + threadIdx.x) >> 5);
  const int qt  = wave % (SEQ / 16);
  const int h   = (wave / (SEQ / 16)) % NH;
  const int b   = wave / ((SEQ / 16) * NH);
  const int kvh = h / (NH / NKV);
  const int q0  = qt * 16;
  const int hi = lane >> 4, lo = lane & 15;

  _Float16* P = smem + wslot * 512;

  // Q fragments (A layout): a0 = dims 0..31, a1 = dims 32..63
  const _Float16* qrow = Qh + (((size_t)(b * SEQ + q0 + lo)) * NH + h) * HD + hi * 8;
  FragA a0, a1;
  a0.h[0] = *(const v8h*)(qrow);      a0.h[1] = *(const v8h*)(qrow + 16);
  a1.h[0] = *(const v8h*)(qrow + 32); a1.h[1] = *(const v8h*)(qrow + 48);

  const v8f vzero = {0.f,0.f,0.f,0.f,0.f,0.f,0.f,0.f};
  v8f o[4];
#pragma unroll
  for (int t = 0; t < 4; ++t) o[t] = vzero;
  float mrow[8], lsum[8];
#pragma unroll
  for (int i = 0; i < 8; ++i) { mrow[i] = -3.0e38f; lsum[i] = 0.f; }

  const int nkb   = (q0 + 16 + 31) >> 5;   // causal: keys 0 .. q0+15
  const int nFull = (q0 + 1) >> 5;         // blocks with all 32 keys <= q0 (no mask)
  const _Float16* kbase  = Kh + ((size_t)b * SEQ * NKV + kvh) * HD + hi * 8;
  const _Float16* vbase0 = Vt + ((size_t)(b * NKV + kvh)) * HD * SEQ + hi * 8;

  for (int kb = 0; kb < nkb; ++kb) {
    const int k0 = kb * 32;
    // K^T B-fragments for two 16-key subtiles
    const _Float16* kr0 = kbase + (size_t)(k0 + lo)      * (NKV * HD);
    const _Float16* kr1 = kbase + (size_t)(k0 + 16 + lo) * (NKV * HD);
    FragA bk0a, bk0b, bk1a, bk1b;
    bk0a.h[0] = *(const v8h*)(kr0);      bk0a.h[1] = *(const v8h*)(kr0 + 16);
    bk0b.h[0] = *(const v8h*)(kr0 + 32); bk0b.h[1] = *(const v8h*)(kr0 + 48);
    bk1a.h[0] = *(const v8h*)(kr1);      bk1a.h[1] = *(const v8h*)(kr1 + 16);
    bk1b.h[0] = *(const v8h*)(kr1 + 32); bk1b.h[1] = *(const v8h*)(kr1 + 48);

    if (kb + 1 < nkb) {  // scalar guard; prefetch next K block (global_prefetch_b8)
      __builtin_prefetch(kbase + (size_t)(k0 + 32 + lo) * (NKV * HD), 0, 3);
      __builtin_prefetch(kbase + (size_t)(k0 + 48 + lo) * (NKV * HD), 0, 3);
    }

    v8f s0 = wmma_f16(a1, bk0b, wmma_f16(a0, bk0a, vzero));
    v8f s1 = wmma_f16(a1, bk1b, wmma_f16(a0, bk1a, vzero));

    float p0[8], p1[8], rm[8];
    if (kb >= nFull) {                     // scalar branch: only diagonal block(s)
#pragma unroll
      for (int i = 0; i < 8; ++i) {
        const int qi = q0 + hi * 8 + i;
        p0[i] = (k0 + lo      <= qi) ? s0[i] : -1.0e9f;
        p1[i] = (k0 + 16 + lo <= qi) ? s1[i] : -1.0e9f;
      }
    } else {
#pragma unroll
      for (int i = 0; i < 8; ++i) { p0[i] = s0[i]; p1[i] = s1[i]; }
    }
#pragma unroll
    for (int i = 0; i < 8; ++i) rm[i] = fmaxf(p0[i], p1[i]);
#pragma unroll
    for (int xm = 1; xm < 16; xm <<= 1)
#pragma unroll
      for (int i = 0; i < 8; ++i)
        rm[i] = fmaxf(rm[i], __shfl_xor(rm[i], xm, 32));

    float corr[8];
#pragma unroll
    for (int i = 0; i < 8; ++i) {
      const float mn = fmaxf(mrow[i], rm[i]);
      corr[i] = __expf(mrow[i] - mn);
      mrow[i] = mn;
      p0[i] = __expf(p0[i] - mn);
      p1[i] = __expf(p1[i] - mn);
    }
    float rs[8];
#pragma unroll
    for (int i = 0; i < 8; ++i) rs[i] = p0[i] + p1[i];
#pragma unroll
    for (int xm = 1; xm < 16; xm <<= 1)
#pragma unroll
      for (int i = 0; i < 8; ++i)
        rs[i] += __shfl_xor(rs[i], xm, 32);
#pragma unroll
    for (int i = 0; i < 8; ++i) lsum[i] = lsum[i] * corr[i] + rs[i];
#pragma unroll
    for (int t = 0; t < 4; ++t)
#pragma unroll
      for (int i = 0; i < 8; ++i) o[t][i] *= corr[i];

    // rotate P from C/D layout to A layout through LDS
#pragma unroll
    for (int i = 0; i < 8; ++i) {
      P[(hi * 8 + i) * 32 + lo]      = (_Float16)p0[i];
      P[(hi * 8 + i) * 32 + 16 + lo] = (_Float16)p1[i];
    }
    asm volatile("s_wait_dscnt 0" ::: "memory");
    FragA pa;
    pa.h[0] = *(const v8h*)(P + lo * 32 + hi * 8);
    pa.h[1] = *(const v8h*)(P + lo * 32 + 16 + hi * 8);
    asm volatile("" ::: "memory");

    // O += P @ V  (Vt is dim-major: contiguous along keys)
    const _Float16* vb = vbase0 + k0;
#pragma unroll
    for (int t = 0; t < 4; ++t) {
      const _Float16* vrow = vb + (size_t)(t * 16 + lo) * SEQ;
      FragA bv;
      bv.h[0] = *(const v8h*)(vrow);
      bv.h[1] = *(const v8h*)(vrow + 16);
      o[t] = wmma_f16(pa, bv, o[t]);
    }
  }

  float inv[8];
#pragma unroll
  for (int i = 0; i < 8; ++i) inv[i] = 1.0f / lsum[i];
#pragma unroll
  for (int t = 0; t < 4; ++t) {
    const int dcol = t * 16 + lo;
#pragma unroll
    for (int i = 0; i < 8; ++i) {
      const int m = q0 + hi * 8 + i;
      Oh[(((size_t)(b * SEQ + m)) * NH + h) * HD + dcol] = (_Float16)(o[t][i] * inv[i]);
    }
  }
}

// ------------------------------------------------------------------ launch
extern "C" void kernel_launch(void* const* d_in, const int* in_sizes, int n_in,
                              void* d_out, int out_size, void* d_ws, size_t ws_size,
                              hipStream_t stream) {
  (void)in_sizes; (void)n_in; (void)out_size; (void)ws_size;
  const float* hidden = (const float*)d_in[0];
  // d_in[1] attention_mask: causal, computed analytically. d_in[2] position_ids: arange.
  const float* Wq = (const float*)d_in[3];
  const float* Wk = (const float*)d_in[4];
  const float* Wv = (const float*)d_in[5];
  const float* Wo = (const float*)d_in[6];

  char* ws = (char*)d_ws;
  size_t off = 0;
  auto alloc = [&](size_t bytes) {
    void* p = ws + off;
    off = (off + bytes + 255) & ~(size_t)255;
    return p;
  };
  _Float16* Xh  = (_Float16*)alloc((size_t)BB * SEQ * HID * 2);
  _Float16* WqT = (_Float16*)alloc((size_t)HID * HID * 2);
  _Float16* WkT = (_Float16*)alloc((size_t)HID * NKV * HD * 2);
  _Float16* WvT = (_Float16*)alloc((size_t)HID * NKV * HD * 2);
  _Float16* WoT = (_Float16*)alloc((size_t)HID * HID * 2);
  _Float16* Qh  = (_Float16*)alloc((size_t)BB * SEQ * NH * HD * 2);
  _Float16* Kh  = (_Float16*)alloc((size_t)BB * SEQ * NKV * HD * 2);
  _Float16* Vt  = (_Float16*)alloc((size_t)BB * NKV * HD * SEQ * 2);
  _Float16* Oh  = (_Float16*)alloc((size_t)BB * SEQ * NH * HD * 2);

  const int M = BB * SEQ;           // 8192

  {
    int n = M * HID;
    cvt_f32_f16<<<(n + 255) / 256, 256, 0, stream>>>(hidden, Xh, n);
  }
  transpose_cvt<<<(HID * HID + 255) / 256,      256, 0, stream>>>(Wq, WqT, HID, HID);
  transpose_cvt<<<(HID * NKV * HD + 255) / 256, 256, 0, stream>>>(Wk, WkT, HID, NKV * HD);
  transpose_cvt<<<(HID * NKV * HD + 255) / 256, 256, 0, stream>>>(Wv, WvT, HID, NKV * HD);
  transpose_cvt<<<(HID * HID + 255) / 256,      256, 0, stream>>>(Wo, WoT, HID, HID);

  {  // Q = X @ Wq   -> Qh [B][S][NH][64] f16
    int waves = (M / 16) * ((NH * HD) / 64);
    gemm_wmma<false, false><<<waves * 32 / 256, 256, 0, stream>>>(Xh, WqT, Qh, M, NH * HD, HID, SEQ);
  }
  {  // K = X @ Wk   -> Kh [B][S][NKV][64] f16
    int waves = (M / 16) * ((NKV * HD) / 64);
    gemm_wmma<false, false><<<waves * 32 / 256, 256, 0, stream>>>(Xh, WkT, Kh, M, NKV * HD, HID, SEQ);
    // V = X @ Wv   -> Vt [B][NKV][64][S] f16 (transposed store)
    gemm_wmma<true,  false><<<waves * 32 / 256, 256, 0, stream>>>(Xh, WvT, Vt, M, NKV * HD, HID, SEQ);
  }
  {  // RoPE in place on Q and K (Q pre-scaled by 1/sqrt(HD))
    int total = BB * SEQ * (NH + NKV) * 32;
    rope_kernel<<<(total + 255) / 256, 256, 0, stream>>>(Qh, Kh, total);
  }
  {  // attention -> Oh [B][S][NH][64] f16
    int waves = BB * NH * (SEQ / 16);
    flash_attn<<<waves * 32 / 256, 256, 0, stream>>>(Qh, Kh, Vt, Oh);
  }
  {  // out = O @ Wo -> f32 d_out [B][S][HID]
    int waves = (M / 16) * (HID / 64);
    gemm_wmma<false, true><<<waves * 32 / 256, 256, 0, stream>>>(Oh, WoT, d_out, M, HID, NH * HD, SEQ);
  }
}
